// Qwen3NextSparseMoeBlock_46145128628424
// MI455X (gfx1250) — matrix operand order repaired and checked by reading, not compiled, not run on hardware
//
#include <hip/hip_runtime.h>
#include <hip/hip_bf16.h>
#include <math.h>

// Problem dims (from reference): T,H,E,K,I,S
#define T_ 512
#define H_ 1024
#define E_ 32
#define K_ 4
#define I_ 512
#define S_ 2048

typedef __attribute__((ext_vector_type(16))) __bf16 v16bf;
typedef __attribute__((ext_vector_type(8)))  __bf16 v8bf;
typedef __attribute__((ext_vector_type(8)))  float  v8f;

// ---- helpers -------------------------------------------------------------

// f32 -> bf16; lowers to v_cvt_pk_bf16_f32 on gfx1250 (verified in disasm).
__device__ __forceinline__ __bf16 f2bf(float f) { return (__bf16)f; }

// silu with fast reciprocal (v_rcp_f32) instead of IEEE div expansion.
__device__ __forceinline__ float silu_f(float x) {
  return x * __builtin_amdgcn_rcpf(1.0f + __expf(-x));
}

__device__ __forceinline__ float sigmoid_f(float x) {
  return __builtin_amdgcn_rcpf(1.0f + __expf(-x));
}

// A-matrix 16x32 bf16 as loaded here (contiguous k, not ISA table order; B matches):
//   lane<16 : row M=lane,    K = 0..15
//   lane>=16: row M=lane-16, K = 16..31
// row_base points at this lane's matrix row (bf16, contiguous K).
__device__ __forceinline__ v16bf load_a_tile(const __bf16* row_base, int k0, int hi) {
  const v8bf lo = *(const v8bf*)(row_base + k0 + hi * 16);
  const v8bf hh = *(const v8bf*)(row_base + k0 + hi * 16 + 8);
  v16bf a;
#pragma unroll
  for (int i = 0; i < 8; ++i) { a[i] = lo[i]; a[8 + i] = hh[i]; }
  return a;
}

// B-matrix 32x16 bf16 as loaded here (same contiguous k order as A):
//   lane<16 : col N=lane,    K = 0..15
//   lane>=16: col N=lane-16, K = 16..31
// B is fp32 row-major [K x ldb]; convert in-register (v_cvt_pk_bf16_f32).
__device__ __forceinline__ v16bf load_b_tile_f32(const float* __restrict__ B,
                                                 int ldb, int kb, int col) {
  v16bf b;
#pragma unroll
  for (int i = 0; i < 16; ++i) b[i] = f2bf(B[(size_t)(kb + i) * ldb + col]);
  return b;
}

__device__ __forceinline__ v8f wmma_bf16(v16bf a, v16bf b, v8f c) {
  // D = A(16x32 bf16) * B(32x16 bf16) + C(16x16 f32)
  return __builtin_amdgcn_wmma_f32_16x16x32_bf16(
      /*neg_a=*/false, a, /*neg_b=*/false, b,
      /*c_mod=*/(short)0, c, /*reuse_a=*/false, /*reuse_b=*/false);
}

// ---- kernel 1: router (logits, softmax, top-4, sigmoid gate) + x->bf16 ----

__global__ void router_kernel(const float* __restrict__ x,
                              const float* __restrict__ Wg,
                              float* __restrict__ sig,        // [T]
                              int*   __restrict__ topk_idx,   // [T*K]
                              float* __restrict__ topk_w,     // [T*K]
                              __bf16* __restrict__ xb16) {    // [T*H]
  __shared__ float xs[H_];
  __shared__ float lg[E_ + 1];
  const int t = blockIdx.x;
  const int tid = threadIdx.x;

  for (int i = tid; i < H_; i += blockDim.x) {
    float v = x[(size_t)t * H_ + i];
    xs[i] = v;
    xb16[(size_t)t * H_ + i] = f2bf(v);
  }
  __syncthreads();

  if (tid < E_ + 1) {
    float acc = 0.0f;
    for (int h = 0; h < H_; ++h) acc = fmaf(xs[h], Wg[(size_t)h * (E_ + 1) + tid], acc);
    lg[tid] = acc;
  }
  __syncthreads();

  if (tid == 0) {
    // shared-expert sigmoid gate
    sig[t] = sigmoid_f(lg[E_]);
    // softmax over router logits (unnormalized exp is enough: top-k
    // renormalization cancels the partition function)
    float mx = lg[0];
    for (int e = 1; e < E_; ++e) mx = fmaxf(mx, lg[e]);
    for (int e = 0; e < E_; ++e) lg[e] = __expf(lg[e] - mx);
    // top-4 (first index wins ties, matching lax.top_k)
    float w[K_]; int idx[K_]; float tot = 0.0f;
#pragma unroll
    for (int k = 0; k < K_; ++k) {
      int best = 0; float bv = -1.0f;
      for (int e = 0; e < E_; ++e) {
        float p = lg[e];
        if (p > bv) { bv = p; best = e; }
      }
      lg[best] = -1.0f;
      idx[k] = best; w[k] = bv; tot += bv;
    }
    float inv = __builtin_amdgcn_rcpf(tot);
#pragma unroll
    for (int k = 0; k < K_; ++k) {
      topk_idx[t * K_ + k] = idx[k];
      topk_w[t * K_ + k]   = w[k] * inv;
    }
  }
}

// ---- kernel 2: deterministic token->expert compaction (no atomics) -------

__global__ void assign_kernel(const int* __restrict__ topk_idx,
                              int* __restrict__ counts,     // [E]
                              int* __restrict__ etok,       // [E*T]
                              int* __restrict__ slot_of) {  // [T*K]
  const int e = threadIdx.x;  // 32 threads, one per expert
  if (e >= E_) return;
  int c = 0;
  for (int t = 0; t < T_; ++t) {
#pragma unroll
    for (int k = 0; k < K_; ++k) {
      if (topk_idx[t * K_ + k] == e) {
        etok[e * T_ + c] = t;
        slot_of[t * K_ + k] = c;
        ++c;
      }
    }
  }
  counts[e] = c;
}

// ---- kernel 3: per-expert gate/up GEMM + SwiGLU -> bf16 h buffer ---------
// grid = (E, I/64), block = 128 (4 waves x 16 output cols; 2 WMMA per A tile)

__global__ void expert_mlp1_kernel(const __bf16* __restrict__ xb16,
                                   const float* __restrict__ Wgate,  // [E,H,I]
                                   const float* __restrict__ Wup,    // [E,H,I]
                                   const int* __restrict__ counts,
                                   const int* __restrict__ etok,
                                   __bf16* __restrict__ hbuf) {      // [E,T,I]
  const int e    = blockIdx.x;
  const int wave = threadIdx.x >> 5;
  const int lane = threadIdx.x & 31;
  const int r    = lane & 15;
  const int hi   = lane >> 4;
  const int col  = blockIdx.y * 64 + wave * 16 + r;
  const int nt   = counts[e];
  const float* Bg = Wgate + (size_t)e * H_ * I_;
  const float* Bu = Wup   + (size_t)e * H_ * I_;
  const int* toks = etok + e * T_;

  // NOTE: for every m0 in this loop, m0+15 <= T_-1 (m0 < nt <= T_, m0 % 16 == 0),
  // so unconditional 16-row stores are always in-bounds; rows >= nt hold
  // garbage that is never consumed downstream.
  for (int m0 = 0; m0 < nt; m0 += 16) {
    const int tr  = m0 + r;
    const int tok = toks[tr < nt ? tr : 0];  // clamp tail rows to a valid token
    const __bf16* arow = xb16 + (size_t)tok * H_;
    v8f accg = {}; v8f accu = {};
    for (int k0 = 0; k0 < H_; k0 += 32) {
      const int kb = k0 + hi * 16;
      if (k0 + 32 < H_) {  // uniform branch: keep the load pipe ahead
        __builtin_prefetch(&Bg[(size_t)(kb + 32) * I_ + col], 0, 1);
        __builtin_prefetch(&Bu[(size_t)(kb + 32) * I_ + col], 0, 1);
      }
      v16bf a  = load_a_tile(arow, k0, hi);
      v16bf bg = load_b_tile_f32(Bg, I_, kb, col);
      v16bf bu = load_b_tile_f32(Bu, I_, kb, col);
      accg = wmma_bf16(a, bg, accg);
      accu = wmma_bf16(a, bu, accu);
    }
#pragma unroll
    for (int j = 0; j < 8; ++j) {
      const int row = m0 + hi * 8 + j;
      hbuf[((size_t)e * T_ + row) * I_ + col] = f2bf(silu_f(accg[j]) * accu[j]);
    }
  }
}

// ---- kernel 4: per-expert down GEMM -> fp32 y buffer ---------------------
// grid = (E, H/128), block = 128 (4 waves x 32 output cols; A reused by 2 WMMAs)

__global__ void expert_mlp2_kernel(const __bf16* __restrict__ hbuf,  // [E,T,I]
                                   const float* __restrict__ Wdown,  // [E,I,H]
                                   const int* __restrict__ counts,
                                   float* __restrict__ ybuf) {       // [E,T,H]
  const int e    = blockIdx.x;
  const int wave = threadIdx.x >> 5;
  const int lane = threadIdx.x & 31;
  const int r    = lane & 15;
  const int hi   = lane >> 4;
  const int col  = blockIdx.y * 128 + wave * 32 + r;  // covers [col, col+16]
  const int nt   = counts[e];
  const float* Bd = Wdown + (size_t)e * I_ * H_;

  for (int m0 = 0; m0 < nt; m0 += 16) {
    const __bf16* arow = hbuf + ((size_t)e * T_ + m0 + r) * I_;  // in-bounds even past nt
    v8f acc0 = {}; v8f acc1 = {};
    for (int k0 = 0; k0 < I_; k0 += 32) {
      const int kb = k0 + hi * 16;
      if (k0 + 32 < I_) {
        __builtin_prefetch(&Bd[(size_t)(kb + 32) * H_ + col], 0, 1);
        __builtin_prefetch(&Bd[(size_t)(kb + 32) * H_ + col + 16], 0, 1);
      }
      v16bf a  = load_a_tile(arow, k0, hi);
      v16bf b0 = load_b_tile_f32(Bd, H_, kb, col);
      v16bf b1 = load_b_tile_f32(Bd, H_, kb, col + 16);
      acc0 = wmma_bf16(a, b0, acc0);
      acc1 = wmma_bf16(a, b1, acc1);
    }
#pragma unroll
    for (int j = 0; j < 8; ++j) {
      const int row = m0 + hi * 8 + j;  // always <= T_-1; extra rows unused
      ybuf[((size_t)e * T_ + row) * H_ + col]      = acc0[j];
      ybuf[((size_t)e * T_ + row) * H_ + col + 16] = acc1[j];
    }
  }
}

// ---- kernel 5: shared-expert gate/up GEMM + SwiGLU -----------------------
// grid = (T/16, S/64), block = 128

__global__ void shared_mlp1_kernel(const __bf16* __restrict__ xb16,
                                   const float* __restrict__ Wsg,  // [H,S]
                                   const float* __restrict__ Wsu,  // [H,S]
                                   __bf16* __restrict__ shbuf) {   // [T,S]
  const int wave = threadIdx.x >> 5;
  const int lane = threadIdx.x & 31;
  const int r    = lane & 15;
  const int hi   = lane >> 4;
  const int m0   = blockIdx.x * 16;
  const int col  = blockIdx.y * 64 + wave * 16 + r;
  const __bf16* arow = xb16 + (size_t)(m0 + r) * H_;

  v8f accg = {}; v8f accu = {};
  for (int k0 = 0; k0 < H_; k0 += 32) {
    const int kb = k0 + hi * 16;
    if (k0 + 32 < H_) {
      __builtin_prefetch(&Wsg[(size_t)(kb + 32) * S_ + col], 0, 1);
      __builtin_prefetch(&Wsu[(size_t)(kb + 32) * S_ + col], 0, 1);
    }
    v16bf a  = load_a_tile(arow, k0, hi);
    v16bf bg = load_b_tile_f32(Wsg, S_, kb, col);
    v16bf bu = load_b_tile_f32(Wsu, S_, kb, col);
    accg = wmma_bf16(a, bg, accg);
    accu = wmma_bf16(a, bu, accu);
  }
#pragma unroll
  for (int j = 0; j < 8; ++j) {
    const int row = m0 + hi * 8 + j;
    shbuf[(size_t)row * S_ + col] = f2bf(silu_f(accg[j]) * accu[j]);
  }
}

// ---- kernel 6: shared down GEMM fused with gated combine -> d_out --------
// grid = (T/16, H/128), block = 128 (4 waves x 32 output cols)

__global__ void shared_mlp2_combine_kernel(const __bf16* __restrict__ shbuf, // [T,S]
                                           const float* __restrict__ Wsd,    // [S,H]
                                           const float* __restrict__ sig,
                                           const int*   __restrict__ topk_idx,
                                           const float* __restrict__ topk_w,
                                           const int*   __restrict__ slot_of,
                                           const float* __restrict__ ybuf,   // [E,T,H]
                                           float* __restrict__ out) {        // [T,H]
  const int wave = threadIdx.x >> 5;
  const int lane = threadIdx.x & 31;
  const int r    = lane & 15;
  const int hi   = lane >> 4;
  const int m0   = blockIdx.x * 16;
  const int col  = blockIdx.y * 128 + wave * 32 + r;  // covers [col, col+16]
  const __bf16* arow = shbuf + (size_t)(m0 + r) * S_;

  v8f acc0 = {}; v8f acc1 = {};
  for (int k0 = 0; k0 < S_; k0 += 32) {
    const int kb = k0 + hi * 16;
    if (k0 + 32 < S_) {
      __builtin_prefetch(&Wsd[(size_t)(kb + 32) * H_ + col], 0, 1);
      __builtin_prefetch(&Wsd[(size_t)(kb + 32) * H_ + col + 16], 0, 1);
    }
    v16bf a  = load_a_tile(arow, k0, hi);
    v16bf b0 = load_b_tile_f32(Wsd, H_, kb, col);
    v16bf b1 = load_b_tile_f32(Wsd, H_, kb, col + 16);
    acc0 = wmma_bf16(a, b0, acc0);
    acc1 = wmma_bf16(a, b1, acc1);
  }
#pragma unroll
  for (int j = 0; j < 8; ++j) {
    const int t = m0 + hi * 8 + j;
    const float sg = sig[t];
    float val0 = sg * acc0[j];                // gated shared expert
    float val1 = sg * acc1[j];
#pragma unroll
    for (int k = 0; k < K_; ++k) {            // routed combine, fixed order
      const int   e = topk_idx[t * K_ + k];
      const int   s = slot_of[t * K_ + k];
      const float w = topk_w[t * K_ + k];
      const float* yrow = ybuf + ((size_t)e * T_ + s) * H_;
      val0 = fmaf(w, yrow[col], val0);
      val1 = fmaf(w, yrow[col + 16], val1);
    }
    out[(size_t)t * H_ + col]      = val0;
    out[(size_t)t * H_ + col + 16] = val1;
  }
}

// ---- launcher ------------------------------------------------------------

extern "C" void kernel_launch(void* const* d_in, const int* in_sizes, int n_in,
                              void* d_out, int out_size, void* d_ws, size_t ws_size,
                              hipStream_t stream) {
  const float* x     = (const float*)d_in[0];
  const float* Wg    = (const float*)d_in[1];
  const float* Wgate = (const float*)d_in[2];
  const float* Wup   = (const float*)d_in[3];
  const float* Wdown = (const float*)d_in[4];
  const float* Wsg   = (const float*)d_in[5];
  const float* Wsu   = (const float*)d_in[6];
  const float* Wsd   = (const float*)d_in[7];
  float* out = (float*)d_out;

  char* p = (char*)d_ws;
  auto alloc = [&](size_t bytes) -> char* {
    char* q = p;
    p += (bytes + 255) & ~(size_t)255;
    return q;
  };
  int*    counts   = (int*)   alloc((size_t)E_ * 4);
  int*    topk_idx = (int*)   alloc((size_t)T_ * K_ * 4);
  int*    slot_of  = (int*)   alloc((size_t)T_ * K_ * 4);
  float*  topk_w   = (float*) alloc((size_t)T_ * K_ * 4);
  float*  sig      = (float*) alloc((size_t)T_ * 4);
  int*    etok     = (int*)   alloc((size_t)E_ * T_ * 4);
  __bf16* xb16     = (__bf16*)alloc((size_t)T_ * H_ * 2);
  __bf16* shbuf    = (__bf16*)alloc((size_t)T_ * S_ * 2);
  __bf16* hbuf     = (__bf16*)alloc((size_t)E_ * T_ * I_ * 2);
  float*  ybuf     = (float*) alloc((size_t)E_ * T_ * H_ * 4);

  router_kernel<<<T_, 128, 0, stream>>>(x, Wg, sig, topk_idx, topk_w, xb16);
  assign_kernel<<<1, 32, 0, stream>>>(topk_idx, counts, etok, slot_of);
  expert_mlp1_kernel<<<dim3(E_, I_ / 64), 128, 0, stream>>>(xb16, Wgate, Wup, counts, etok, hbuf);
  expert_mlp2_kernel<<<dim3(E_, H_ / 128), 128, 0, stream>>>(hbuf, Wdown, counts, ybuf);
  shared_mlp1_kernel<<<dim3(T_ / 16, S_ / 64), 128, 0, stream>>>(xb16, Wsg, Wsu, shbuf);
  shared_mlp2_combine_kernel<<<dim3(T_ / 16, H_ / 128), 128, 0, stream>>>(
      shbuf, Wsd, sig, topk_idx, topk_w, slot_of, ybuf, out);
}
